// DGCNN_34900904248087
// MI455X (gfx1250) — matrix operand order repaired and checked by reading, not compiled, not run on hardware
//
#include <hip/hip_runtime.h>
#include <hip/hip_bf16.h>

typedef _Float16 half_t;
typedef __attribute__((ext_vector_type(16))) _Float16 v16h;
typedef __attribute__((ext_vector_type(8)))  _Float16 v8h;
typedef __attribute__((ext_vector_type(8)))  float    v8f;

#define BATCH 8
#define NPTS  2048
#define KNN   20
#define MROWS (BATCH * NPTS * KNN)   // 327680
#define M5    (BATCH * NPTS)         // 16384

// ---------------------------------------------------------------------------
// Zero fill (f32)
// ---------------------------------------------------------------------------
__global__ void zerof_kernel(float* __restrict__ p, size_t n) {
    size_t i = (size_t)blockIdx.x * blockDim.x + threadIdx.x;
    size_t stride = (size_t)gridDim.x * blockDim.x;
    for (; i < n; i += stride) p[i] = 0.0f;
}

// ---------------------------------------------------------------------------
// kNN: one wave (32 lanes) per query point. Distances staged in LDS,
// 20 rounds of (min, argmin) wave reduction with smallest-index tie break.
// ---------------------------------------------------------------------------
__global__ __launch_bounds__(128) void knn_kernel(const float* __restrict__ x,
                                                  int* __restrict__ idx) {
    __shared__ float sD[4][NPTS];   // 32 KB
    const int lane = threadIdx.x & 31;
    const int w    = threadIdx.x >> 5;
    const int qi   = blockIdx.x * 4 + w;       // 0 .. B*N-1
    const int b    = qi >> 11;
    const int n    = qi & (NPTS - 1);
    const float* xb = x + (size_t)b * 3 * NPTS;
    const float q0 = xb[n], q1 = xb[NPTS + n], q2 = xb[2 * NPTS + n];
    for (int j = lane; j < NPTS; j += 32) {
        float d0 = xb[j] - q0;
        float d1 = xb[NPTS + j] - q1;
        float d2 = xb[2 * NPTS + j] - q2;
        sD[w][j] = d0 * d0 + d1 * d1 + d2 * d2;
    }
    __syncthreads();
    int* orow = idx + (size_t)qi * KNN;
    for (int r = 0; r < KNN; ++r) {
        float bd = 3.4e38f; int bi = 0x7fffffff;
        for (int j = lane; j < NPTS; j += 32) {
            float d = sD[w][j];
            if (d < bd || (d == bd && j < bi)) { bd = d; bi = j; }
        }
        for (int off = 16; off; off >>= 1) {
            float od = __shfl_down(bd, off, 32);
            int   oi = __shfl_down(bi, off, 32);
            if (od < bd || (od == bd && oi < bi)) { bd = od; bi = oi; }
        }
        if (lane == 0) { orow[r] = bi; sD[w][bi] = 3.4e38f; }
        __syncthreads();
    }
}

// ---------------------------------------------------------------------------
// Gather: build f16 feature matrix a0[M, 32] = [nbr(3), ctr(3), 0...]
// ---------------------------------------------------------------------------
__global__ void gather_kernel(const float* __restrict__ x,
                              const int* __restrict__ idx,
                              half_t* __restrict__ a0) {
    int m = blockIdx.x * 256 + threadIdx.x;
    if (m >= MROWS) return;
    int bn = m / KNN;
    int n  = bn & (NPTS - 1);
    int b  = bn >> 11;
    int j  = idx[m];
    const float* xb = x + (size_t)b * 3 * NPTS;
    v16h v = {};
    v[0] = (half_t)xb[j];            v[1] = (half_t)xb[NPTS + j];
    v[2] = (half_t)xb[2 * NPTS + j];
    v[3] = (half_t)xb[n];            v[4] = (half_t)xb[NPTS + n];
    v[5] = (half_t)xb[2 * NPTS + n];
    v16h z = {};
    v16h* dst = (v16h*)(a0 + (size_t)m * 32);
    dst[0] = v;
    dst[1] = z;
}

// ---------------------------------------------------------------------------
// Weight conversion: f32 [O, Cin] -> f16 [O, Cpad] zero-padded
// ---------------------------------------------------------------------------
__global__ void convw_kernel(const float* __restrict__ W, half_t* __restrict__ Wh,
                             int O, int Cin, int Cpad) {
    int t = blockIdx.x * 256 + threadIdx.x;
    if (t >= O * Cpad) return;
    int o = t / Cpad, c = t % Cpad;
    Wh[t] = (c < Cin) ? (half_t)W[o * Cin + c] : (half_t)0.0f;
}

// ---------------------------------------------------------------------------
// WMMA GEMM: Y[M,O] (f32) = A[M,C] (f16) x W[O,C]^T (f16)
// 8 waves / block; each wave owns a 16-row M tile and sweeps O in blocks of
// 64 channels (4 WMMA tiles) per K pass so A fragments are loaded once per
// 4 WMMAs. Fragments use contiguous K order 16*hi+j in A and B (wave32, 16x16x32 f16).
// Requires O % 64 == 0, C % 32 == 0.
// ---------------------------------------------------------------------------
__global__ __launch_bounds__(256) void gemm_wmma_kernel(
    const half_t* __restrict__ A, const half_t* __restrict__ W,
    float* __restrict__ Y, int M, int C, int O) {
    const int lane  = threadIdx.x & 31;
    const int wave  = threadIdx.x >> 5;
    const int mbase = blockIdx.x * 128 + wave * 16;
    if (mbase >= M) return;                       // wave-uniform
    const int r  = lane & 15;
    const int hi = lane >> 4;
    const half_t* arow = A + (size_t)(mbase + r) * C;
    for (int ot = 0; ot < O; ot += 64) {
        v8f acc0 = {}, acc1 = {}, acc2 = {}, acc3 = {};
        const half_t* wrow0 = W + (size_t)(ot + r) * C;
        const half_t* wrow1 = wrow0 + (size_t)16 * C;
        const half_t* wrow2 = wrow0 + (size_t)32 * C;
        const half_t* wrow3 = wrow0 + (size_t)48 * C;
        for (int kc = 0; kc < C; kc += 32) {
            // A 16x32 frag: elems 0..15 = K kc+16*hi+j
            v8h alo = *(const v8h*)(arow + kc + 16 * hi);
            v8h ahi = *(const v8h*)(arow + kc + 16 * hi + 8);
            v16h a = __builtin_shufflevector(alo, ahi,
                0, 1, 2, 3, 4, 5, 6, 7, 8, 9, 10, 11, 12, 13, 14, 15);
            // B 32x16 frags: lane col = r, K run = kc+16*hi .. +15
            v16h b0 = *(const v16h*)(wrow0 + kc + 16 * hi);
            v16h b1 = *(const v16h*)(wrow1 + kc + 16 * hi);
            v16h b2 = *(const v16h*)(wrow2 + kc + 16 * hi);
            v16h b3 = *(const v16h*)(wrow3 + kc + 16 * hi);
            if (kc + 32 < C)
                __builtin_prefetch(arow + kc + 32 + 8 * hi, 0, 1);
            acc0 = __builtin_amdgcn_wmma_f32_16x16x32_f16(
                false, a, false, b0, (short)0, acc0, false, false);
            acc1 = __builtin_amdgcn_wmma_f32_16x16x32_f16(
                false, a, false, b1, (short)0, acc1, false, false);
            acc2 = __builtin_amdgcn_wmma_f32_16x16x32_f16(
                false, a, false, b2, (short)0, acc2, false, false);
            acc3 = __builtin_amdgcn_wmma_f32_16x16x32_f16(
                false, a, false, b3, (short)0, acc3, false, false);
        }
        // D 16x16 f32: col = r, rows i + 8*hi
        float* ybase = Y + (size_t)(mbase + 8 * hi) * O + ot + r;
        #pragma unroll
        for (int i = 0; i < 8; ++i) {
            float* yrow = ybase + (size_t)i * O;
            yrow[0]  = acc0[i];
            yrow[16] = acc1[i];
            yrow[32] = acc2[i];
            yrow[48] = acc3[i];
        }
    }
}

// ---------------------------------------------------------------------------
// Per-channel sum / sumsq reduction over rows (for batch-norm statistics)
// ---------------------------------------------------------------------------
__global__ void colsum_kernel(const float* __restrict__ Y, int M, int O,
                              float* __restrict__ s1, float* __restrict__ s2) {
    int r0   = blockIdx.x * 1024;
    int rend = r0 + 1024; if (rend > M) rend = M;
    for (int o = threadIdx.x; o < O; o += 256) {
        float s = 0.0f, q = 0.0f;
        for (int r = r0; r < rend; ++r) {
            float v = Y[(size_t)r * O + o];
            s += v; q += v * v;
        }
        atomicAdd(&s1[o], s);
        atomicAdd(&s2[o], q);
    }
}

// ---------------------------------------------------------------------------
// BN (batch stats) + ReLU + (optional) f16 activation write + max over K
// into the 512-channel cat buffer at catOff.
// ---------------------------------------------------------------------------
__global__ void bnmax_kernel(const float* __restrict__ Y,
                             const float* __restrict__ s1, const float* __restrict__ s2,
                             const float* __restrict__ g,  const float* __restrict__ bt,
                             half_t* __restrict__ act, half_t* __restrict__ cat,
                             int catOff, int O, float invM) {
    int tid = blockIdx.x * 256 + threadIdx.x;
    int total = BATCH * NPTS * O;
    if (tid >= total) return;
    int o  = tid % O;
    int bn = tid / O;
    float mean = s1[o] * invM;
    float var  = s2[o] * invM - mean * mean;
    float sc = g[o] * rsqrtf(var + 1e-5f);
    float sh = bt[o] - mean * sc;
    float mx = 0.0f;
    const float* yrow = Y + (size_t)bn * KNN * O + o;
    half_t* arow = act ? (act + (size_t)bn * KNN * O + o) : (half_t*)nullptr;
    for (int k = 0; k < KNN; ++k) {
        float v = yrow[(size_t)k * O] * sc + sh;
        v = v > 0.0f ? v : 0.0f;
        if (arow) arow[(size_t)k * O] = (half_t)v;
        if (v > mx) mx = v;
    }
    cat[(size_t)bn * 512 + catOff + o] = (half_t)mx;
}

// ---------------------------------------------------------------------------
// Final BN+ReLU, write [B, 512, N] into zero-padded [B, 1024, N] output
// ---------------------------------------------------------------------------
__global__ void bnfinal_kernel(const float* __restrict__ Y5,
                               const float* __restrict__ s1, const float* __restrict__ s2,
                               const float* __restrict__ g,  const float* __restrict__ bt,
                               float* __restrict__ out) {
    int tid = blockIdx.x * 256 + threadIdx.x;          // B*512*N threads
    int n = tid & (NPTS - 1);
    int t = tid >> 11;
    int o = t & 511;
    int b = t >> 9;
    const float invM = 1.0f / (float)M5;
    float mean = s1[o] * invM;
    float var  = s2[o] * invM - mean * mean;
    float sc = g[o] * rsqrtf(var + 1e-5f);
    float sh = bt[o] - mean * sc;
    float raw = Y5[((size_t)(b * NPTS + n)) * 512 + o];
    float v = raw * sc + sh;
    v = v > 0.0f ? v : 0.0f;
    out[(size_t)b * 1024 * NPTS + (size_t)o * NPTS + n] = v;
}

// ---------------------------------------------------------------------------
extern "C" void kernel_launch(void* const* d_in, const int* in_sizes, int n_in,
                              void* d_out, int out_size, void* d_ws, size_t ws_size,
                              hipStream_t stream) {
    (void)in_sizes; (void)n_in; (void)out_size; (void)ws_size;
    const float* x = (const float*)d_in[0];
    const float* W[5]  = {(const float*)d_in[1], (const float*)d_in[2],
                          (const float*)d_in[3], (const float*)d_in[4],
                          (const float*)d_in[5]};
    const float* g[5]  = {(const float*)d_in[6],  (const float*)d_in[8],
                          (const float*)d_in[10], (const float*)d_in[12],
                          (const float*)d_in[14]};
    const float* bt[5] = {(const float*)d_in[7],  (const float*)d_in[9],
                          (const float*)d_in[11], (const float*)d_in[13],
                          (const float*)d_in[15]};
    float* out = (float*)d_out;

    // ---- workspace carve (all sizes multiples of 256 bytes) ----
    char* ws = (char*)d_ws;
    size_t off = 0;
    auto take = [&](size_t bytes) { char* p = ws + off; off += (bytes + 255) & ~(size_t)255; return p; };
    int*    idx   = (int*)   take((size_t)MROWS * 4);           // 1.3 MB
    half_t* a0    = (half_t*)take((size_t)MROWS * 32 * 2);      // 21 MB
    half_t* actA  = (half_t*)take((size_t)MROWS * 128 * 2);     // 84 MB (a1, a3)
    half_t* actB  = (half_t*)take((size_t)MROWS * 64  * 2);     // 42 MB (a2)
    float*  Yraw  = (float*) take((size_t)MROWS * 256 * 4);     // 335 MB (reused)
    half_t* cat   = (half_t*)take((size_t)M5 * 512 * 2);        // 16 MB
    half_t* Wh1   = (half_t*)take((size_t)64  * 32  * 2);
    half_t* Wh2   = (half_t*)take((size_t)64  * 64  * 2);
    half_t* Wh3   = (half_t*)take((size_t)128 * 64  * 2);
    half_t* Wh4   = (half_t*)take((size_t)256 * 128 * 2);
    half_t* Wh5   = (half_t*)take((size_t)512 * 512 * 2);
    float*  stats = (float*) take((size_t)5 * 2 * 512 * 4);     // 20 KB

    // ---- zero output (pad channels) and BN stat accumulators ----
    zerof_kernel<<<4096, 256, 0, stream>>>(out, (size_t)BATCH * 1024 * NPTS);
    zerof_kernel<<<8, 256, 0, stream>>>(stats, (size_t)5 * 2 * 512);

    // ---- kNN + graph feature gather ----
    knn_kernel<<<(BATCH * NPTS) / 4, 128, 0, stream>>>(x, idx);
    gather_kernel<<<(MROWS + 255) / 256, 256, 0, stream>>>(x, idx, a0);

    // ---- convert weights to f16 (Cpad = K-dim padded to multiple of 32) ----
    const int Oc[5]   = {64, 64, 128, 256, 512};
    const int Cin[5]  = {6, 64, 64, 128, 512};
    const int Cpad[5] = {32, 64, 64, 128, 512};
    half_t* Wh[5] = {Wh1, Wh2, Wh3, Wh4, Wh5};
    for (int l = 0; l < 5; ++l) {
        int tot = Oc[l] * Cpad[l];
        convw_kernel<<<(tot + 255) / 256, 256, 0, stream>>>(W[l], Wh[l], Oc[l], Cin[l], Cpad[l]);
    }

    const float invM = 1.0f / (float)MROWS;
    half_t* acts[4]   = {a0, actA, actB, actA};      // inputs to layers 1..4
    half_t* outs[4]   = {actA, actB, actA, nullptr}; // f16 activation outputs
    int     catoff[4] = {0, 64, 128, 256};

    // ---- EdgeConv layers 1..4: GEMM -> stats -> BN/ReLU/max ----
    for (int l = 0; l < 4; ++l) {
        float* s1 = stats + l * 1024;
        float* s2 = s1 + 512;
        gemm_wmma_kernel<<<MROWS / 128, 256, 0, stream>>>(
            acts[l], Wh[l], Yraw, MROWS, Cpad[l], Oc[l]);
        colsum_kernel<<<(MROWS + 1023) / 1024, 256, 0, stream>>>(
            Yraw, MROWS, Oc[l], s1, s2);
        int tot = BATCH * NPTS * Oc[l];
        bnmax_kernel<<<(tot + 255) / 256, 256, 0, stream>>>(
            Yraw, s1, s2, g[l], bt[l], outs[l], cat, catoff[l], Oc[l], invM);
    }

    // ---- Layer 5: [M5,512] x [512,512] -> BN/ReLU -> padded output ----
    {
        float* s1 = stats + 4 * 1024;
        float* s2 = s1 + 512;
        gemm_wmma_kernel<<<M5 / 128, 256, 0, stream>>>(cat, Wh5, Yraw, M5, 512, 512);
        colsum_kernel<<<(M5 + 1023) / 1024, 256, 0, stream>>>(Yraw, M5, 512, s1, s2);
        int tot = BATCH * 512 * NPTS;
        bnfinal_kernel<<<(tot + 255) / 256, 256, 0, stream>>>(Yraw, s1, s2, g[4], bt[4], out);
    }
}
